// TransformerDecoder_7146825580791
// MI455X (gfx1250) — matrix operand order repaired and checked by reading, not compiled, not run on hardware
//
#include <hip/hip_runtime.h>
#include <hip/hip_bf16.h>
#include <stdint.h>

// Problem constants (match reference)
static constexpr int VV  = 32000;
static constexpr int DD  = 512;
static constexpr int HH  = 8;
static constexpr int LL  = 2;
static constexpr int FF  = 2048;   // DFF
static constexpr int BB  = 4;
static constexpr int TT  = 256;
static constexpr int DK  = 64;
static constexpr int MM  = BB * TT; // 1024 rows

typedef __bf16 bf16;
typedef __attribute__((ext_vector_type(16))) __bf16   v16bf;
typedef __attribute__((ext_vector_type(8)))  float    v8f;
typedef __attribute__((ext_vector_type(4)))  uint32_t v4u;

// ---------------------------------------------------------------------------
// WMMA GEMM:  C[m,n] = sum_k A[m,k] * Bt[n,k]   (+bias, +relu)
// A:  bf16 row-major [M x K], leading dim lda
// Bt: bf16 "B transposed" [N x K], leading dim ldb (k-pairs contiguous)
// Batched over grid.y: off = (z/inner)*Out + (z%inner)*In for A/B/C.
// Register-blocked: each wave computes a 32x64 block (2 M-tiles x 4 N-tiles,
// 8 f32 accumulators) -> 8 WMMAs per 12 b128 loads per K-step (24 B per lane per WMMA).
// Requires: M % 32 == 0, N % 64 == 0, K % 32 == 0.
// ---------------------------------------------------------------------------
__global__ __launch_bounds__(256) void k_gemm(
    const bf16* __restrict__ A, int lda, int aOut, int aIn,
    const bf16* __restrict__ Bt, int ldb, int bOut, int bIn,
    float* __restrict__ C, int ldc, int cOut, int cIn,
    int M, int N, int K, int inner,
    const float* __restrict__ bias, int flags)
{
    const int z  = blockIdx.y;
    const int zo = z / inner, zi = z % inner;
    const bf16* Ab = A  + (size_t)zo * aOut + (size_t)zi * aIn;
    const bf16* Bb = Bt + (size_t)zo * bOut + (size_t)zi * bIn;
    float*      Cb = C  + (size_t)zo * cOut + (size_t)zi * cIn;

    const int nBn    = N >> 6;                 // 64-wide N blocks
    const int blocks = (M >> 5) * nBn;         // 32-tall M blocks
    const int wave   = blockIdx.x * (blockDim.x >> 5) + (threadIdx.x >> 5);
    if (wave >= blocks) return;                // wave-uniform: EXEC stays all-1s
    const int bm = wave / nBn, bn = wave % nBn;

    const int lane = threadIdx.x & 31;
    const int lrow = lane & 15;                // M (for A) / N (for B/C) index
    const int lhi  = lane >> 4;                // lane-half select

    const bf16* Ap0 = Ab + (size_t)(bm * 32 + lrow) * lda;
    const bf16* Ap1 = Ap0 + (size_t)16 * lda;
    const bf16* Bp  = Bb + (size_t)(bn * 64 + lrow) * ldb;
    const size_t bstep = (size_t)16 * ldb;     // stride between N-tiles of Bt

    v8f c[8];                                  // c[mi*4 + nj]
#pragma unroll
    for (int i = 0; i < 8; ++i)
#pragma unroll
        for (int r = 0; r < 8; ++r) c[i][r] = 0.0f;

    union Frag { v4u q[2]; v16bf v; };

    for (int k0 = 0; k0 < K; k0 += 32) {
        Frag a0, a1, b[4];
        // A 16x32 bf16 fragment (ISA 7.12.2): VGPR v -> K = (v>>2)*16 + lhi*8 + (v&3)*2
        a0.q[0] = *(const v4u*)(Ap0 + k0 + lhi * 8);
        a0.q[1] = *(const v4u*)(Ap0 + k0 + 16 + lhi * 8);
        a1.q[0] = *(const v4u*)(Ap1 + k0 + lhi * 8);
        a1.q[1] = *(const v4u*)(Ap1 + k0 + 16 + lhi * 8);
        // B 32x16 bf16 fragments (column-major load): VGPR v -> K = (v>>2)*16 + lhi*8 + (v&3)*2
#pragma unroll
        for (int j = 0; j < 4; ++j) {
            b[j].q[0] = *(const v4u*)(Bp + (size_t)j * bstep + k0 + lhi * 8);
            b[j].q[1] = *(const v4u*)(Bp + (size_t)j * bstep + k0 + 16 + lhi * 8);
        }
#pragma unroll
        for (int j = 0; j < 4; ++j)
            c[j] = __builtin_amdgcn_wmma_f32_16x16x32_bf16(
                       false, a0.v, false, b[j].v, (short)0, c[j], false, false);
#pragma unroll
        for (int j = 0; j < 4; ++j)
            c[4 + j] = __builtin_amdgcn_wmma_f32_16x16x32_bf16(
                       false, a1.v, false, b[j].v, (short)0, c[4 + j], false, false);
    }

#pragma unroll
    for (int mi = 0; mi < 2; ++mi)
#pragma unroll
        for (int nj = 0; nj < 4; ++nj) {
            const int n = bn * 64 + nj * 16 + lrow;
            const float bv = (flags & 2) ? bias[n] : 0.0f;
            const v8f acc = c[mi * 4 + nj];
#pragma unroll
            for (int r = 0; r < 8; ++r) {
                const int m = bm * 32 + mi * 16 + lhi * 8 + r;  // C/D: VGPR r -> M = lhi*8+r
                float val = acc[r] + bv;
                if (flags & 1) val = fmaxf(val, 0.0f);
                Cb[(size_t)m * ldc + n] = val;
            }
        }
}

// ---------------------------------------------------------------------------
// Elementwise / helper kernels
// ---------------------------------------------------------------------------
__global__ void k_cvt_bf16(const float* __restrict__ in, bf16* __restrict__ out, int n) {
    int i = blockIdx.x * blockDim.x + threadIdx.x;
    if (i < n) out[i] = (bf16)in[i];
}

// Wt[mat][n][k] = W[mat][k][n], converted to bf16. total = nmat*K*N
__global__ void k_tcvt(const float* __restrict__ W, bf16* __restrict__ Wt,
                       int K, int N, int total) {
    int i = blockIdx.x * blockDim.x + threadIdx.x;
    if (i >= total) return;
    int per = K * N;
    int mat = i / per, r = i - mat * per;
    int k = r / N, n = r - k * N;
    Wt[(size_t)mat * per + (size_t)n * K + k] = (bf16)W[i];
}

// vt[b][h][d][t] = v[b][t][h*DK+d]  (per-head transpose to bf16)
__global__ void k_vt(const float* __restrict__ v, bf16* __restrict__ vt) {
    int i = blockIdx.x * blockDim.x + threadIdx.x;   // over B*T*D
    if (i >= MM * DD) return;
    int d = i % DD;
    int t = (i / DD) % TT;
    int b = i / (DD * TT);
    int h = d / DK, dd2 = d % DK;
    vt[(((size_t)(b * HH + h)) * DK + dd2) * TT + t] = (bf16)v[i];
}

// x[b,t,d] = E[tgt[b,t]][d]*sqrt(D) + PE[t][d]
__global__ void k_embed(const int* __restrict__ tgt, const float* __restrict__ E,
                        float* __restrict__ x) {
    int i = blockIdx.x * blockDim.x + threadIdx.x;
    if (i >= MM * DD) return;
    int d  = i % DD;
    int bt = i / DD;
    int t  = bt % TT;
    int tok = tgt[bt];
    float expo = (float)(d & ~1) / (float)DD;
    float ang  = (float)t * powf(10000.0f, -expo);
    float pe   = (d & 1) ? cosf(ang) : sinf(ang);
    x[i] = E[(size_t)tok * DD + d] * 22.62741699796952f + pe;
}

// LayerNorm over D=512 per row; 256 threads per row (2 elems/thread)
__global__ __launch_bounds__(256) void k_ln(const float* __restrict__ x,
                                            const float* __restrict__ g,
                                            const float* __restrict__ b,
                                            float* __restrict__ out) {
    int row = blockIdx.x, t = threadIdx.x;
    const float* xr = x + (size_t)row * DD;
    float a0 = xr[t], a1 = xr[t + 256];
    __shared__ float s1[256], s2[256];
    s1[t] = a0 + a1;
    s2[t] = a0 * a0 + a1 * a1;
    __syncthreads();
    for (int st = 128; st > 0; st >>= 1) {
        if (t < st) { s1[t] += s1[t + st]; s2[t] += s2[t + st]; }
        __syncthreads();
    }
    float mean = s1[0] * (1.0f / DD);
    float var  = s2[0] * (1.0f / DD) - mean * mean;
    float inv  = rsqrtf(var + 1e-6f);
    float* orow = out + (size_t)row * DD;
    orow[t]       = g[t]       * (a0 - mean) * inv + b[t];
    orow[t + 256] = g[t + 256] * (a1 - mean) * inv + b[t + 256];
}

__global__ void k_add(float* __restrict__ x, const float* __restrict__ h, int n) {
    int i = blockIdx.x * blockDim.x + threadIdx.x;
    if (i < n) x[i] += h[i];
}

// masked softmax over rows of length T=256; one block per (b,h,q) row
__global__ __launch_bounds__(256) void k_softmax(float* __restrict__ sc,
                                                 const int* __restrict__ src_len,
                                                 int causal) {
    int row = blockIdx.x;            // over B*H*T
    int q   = row % TT;
    int b   = row / (HH * TT);
    int col = threadIdx.x;
    float s = sc[(size_t)row * TT + col] * 0.125f;   // 1/sqrt(DK)
    bool ok = causal ? (col <= q) : (col < src_len[b]);
    if (!ok) s = -1e9f;
    __shared__ float red[256];
    red[col] = s; __syncthreads();
    for (int st = 128; st > 0; st >>= 1) {
        if (col < st) red[col] = fmaxf(red[col], red[col + st]);
        __syncthreads();
    }
    float mx = red[0]; __syncthreads();
    float e = __expf(s - mx);
    red[col] = e; __syncthreads();
    for (int st = 128; st > 0; st >>= 1) {
        if (col < st) red[col] += red[col + st];
        __syncthreads();
    }
    sc[(size_t)row * TT + col] = e / red[0];
}

// ---------------------------------------------------------------------------
// Host orchestration
// ---------------------------------------------------------------------------
extern "C" void kernel_launch(void* const* d_in, const int* in_sizes, int n_in,
                              void* d_out, int out_size, void* d_ws, size_t ws_size,
                              hipStream_t stream) {
    (void)in_sizes; (void)n_in; (void)out_size; (void)ws_size;
    const int*   tgt      = (const int*)d_in[0];
    const int*   src_len  = (const int*)d_in[1];
    const float* src_enc  = (const float*)d_in[2];
    const float* E        = (const float*)d_in[3];
    const float* self_qkv = (const float*)d_in[4];
    const float* cross_qkv= (const float*)d_in[5];
    const float* W1       = (const float*)d_in[6];
    const float* b1       = (const float*)d_in[7];
    const float* W2       = (const float*)d_in[8];
    const float* b2       = (const float*)d_in[9];
    const float* ln_g     = (const float*)d_in[10];
    const float* ln_b     = (const float*)d_in[11];
    const float* lnf_g    = (const float*)d_in[12];
    const float* lnf_b    = (const float*)d_in[13];
    float* out = (float*)d_out;

    // bump allocator on workspace (hipMalloc base is 256B aligned)
    char* p = (char*)d_ws;
    auto alloc = [&](size_t bytes) -> void* {
        void* r = (void*)p;
        p += (bytes + 255) & ~(size_t)255;
        return r;
    };
    bf16*  Eb     = (bf16*)alloc((size_t)VV * DD * 2);
    bf16*  sWt    = (bf16*)alloc((size_t)LL * 4 * DD * DD * 2);
    bf16*  cWt    = (bf16*)alloc((size_t)LL * 4 * DD * DD * 2);
    bf16*  W1t    = (bf16*)alloc((size_t)LL * FF * DD * 2);
    bf16*  W2t    = (bf16*)alloc((size_t)LL * DD * FF * 2);
    bf16*  srcb   = (bf16*)alloc((size_t)MM * DD * 2);
    float* x      = (float*)alloc((size_t)MM * DD * 4);
    float* lnf32  = (float*)alloc((size_t)MM * DD * 4);
    bf16*  lnb    = (bf16*)alloc((size_t)MM * DD * 2);
    float* qf     = (float*)alloc((size_t)MM * DD * 4);
    float* kf     = (float*)alloc((size_t)MM * DD * 4);
    float* vf     = (float*)alloc((size_t)MM * DD * 4);
    float* of     = (float*)alloc((size_t)MM * DD * 4);
    float* hf     = (float*)alloc((size_t)MM * DD * 4);
    bf16*  qb     = (bf16*)alloc((size_t)MM * DD * 2);
    bf16*  kb     = (bf16*)alloc((size_t)MM * DD * 2);
    bf16*  vtb    = (bf16*)alloc((size_t)MM * DD * 2);
    bf16*  ob     = (bf16*)alloc((size_t)MM * DD * 2);
    float* scores = (float*)alloc((size_t)BB * HH * TT * TT * 4);
    bf16*  attnb  = (bf16*)alloc((size_t)BB * HH * TT * TT * 2);
    float* ffh    = (float*)alloc((size_t)MM * FF * 4);
    bf16*  ffhb   = (bf16*)alloc((size_t)MM * FF * 2);

    auto cvt = [&](const float* src, bf16* dst, int n) {
        k_cvt_bf16<<<(n + 255) / 256, 256, 0, stream>>>(src, dst, n);
    };
    auto gemm = [&](const bf16* A, int lda, int aO, int aI,
                    const bf16* Bt, int ldb, int bO, int bI,
                    float* C, int ldc, int cO, int cI,
                    int M, int N, int K, int inner, int batch,
                    const float* bias, int flags) {
        int blocks = (M / 32) * (N / 64);      // one 32x64 block per wave
        dim3 grid((blocks + 7) / 8, batch, 1);
        k_gemm<<<grid, 256, 0, stream>>>(A, lda, aO, aI, Bt, ldb, bO, bI,
                                         C, ldc, cO, cI, M, N, K, inner, bias, flags);
    };

    // one-time conversions (per launch; deterministic)
    { int tot = LL * 4 * DD * DD;
      k_tcvt<<<(tot + 255) / 256, 256, 0, stream>>>(self_qkv,  sWt, DD, DD, tot);
      k_tcvt<<<(tot + 255) / 256, 256, 0, stream>>>(cross_qkv, cWt, DD, DD, tot); }
    { int tot = LL * DD * FF;
      k_tcvt<<<(tot + 255) / 256, 256, 0, stream>>>(W1, W1t, DD, FF, tot); }
    { int tot = LL * FF * DD;
      k_tcvt<<<(tot + 255) / 256, 256, 0, stream>>>(W2, W2t, FF, DD, tot); }
    cvt(E, Eb, VV * DD);          // logits weight: Bt[v][k] == E[v][k]
    cvt(src_enc, srcb, MM * DD);
    k_embed<<<(MM * DD + 255) / 256, 256, 0, stream>>>(tgt, E, x);

    auto attention = [&](const bf16* Wt4, const bf16* kvb, int causal) {
        gemm(lnb, DD, 0, 0, Wt4 + 0 * DD * DD, DD, 0, 0, qf, DD, 0, 0, MM, DD, DD, 1, 1, nullptr, 0);
        gemm(kvb, DD, 0, 0, Wt4 + 1 * DD * DD, DD, 0, 0, kf, DD, 0, 0, MM, DD, DD, 1, 1, nullptr, 0);
        gemm(kvb, DD, 0, 0, Wt4 + 2 * DD * DD, DD, 0, 0, vf, DD, 0, 0, MM, DD, DD, 1, 1, nullptr, 0);
        cvt(qf, qb, MM * DD);
        cvt(kf, kb, MM * DD);
        k_vt<<<(MM * DD + 255) / 256, 256, 0, stream>>>(vf, vtb);
        // scores[b,h,q,k] = q . k   (batched over B*H; per-batch offset b*T*D + h*DK)
        gemm(qb, DD, TT * DD, DK, kb, DD, TT * DD, DK,
             scores, TT, HH * TT * TT, TT * TT,
             TT, TT, DK, HH, BB * HH, nullptr, 0);
        k_softmax<<<BB * HH * TT, 256, 0, stream>>>(scores, src_len, causal);
        cvt(scores, attnb, BB * HH * TT * TT);
        // o[b,t,h*DK+d] = sum_k attn[b,h,t,k] * v[b,k,h*DK+d]
        gemm(attnb, TT, HH * TT * TT, TT * TT, vtb, TT, HH * DK * TT, DK * TT,
             of, DD, TT * DD, DK,
             TT, DK, TT, HH, BB * HH, nullptr, 0);
        cvt(of, ob, MM * DD);
        gemm(ob, DD, 0, 0, Wt4 + 3 * DD * DD, DD, 0, 0, hf, DD, 0, 0, MM, DD, DD, 1, 1, nullptr, 0);
        k_add<<<(MM * DD + 255) / 256, 256, 0, stream>>>(x, hf, MM * DD);
    };

    for (int l = 0; l < LL; ++l) {
        // self-attention (causal)
        k_ln<<<MM, 256, 0, stream>>>(x, ln_g + (l * 3 + 0) * DD, ln_b + (l * 3 + 0) * DD, lnf32);
        cvt(lnf32, lnb, MM * DD);
        attention(sWt + (size_t)l * 4 * DD * DD, lnb, 1);
        // cross-attention (src_len mask)
        k_ln<<<MM, 256, 0, stream>>>(x, ln_g + (l * 3 + 1) * DD, ln_b + (l * 3 + 1) * DD, lnf32);
        cvt(lnf32, lnb, MM * DD);
        attention(cWt + (size_t)l * 4 * DD * DD, srcb, 0);
        // FFN
        k_ln<<<MM, 256, 0, stream>>>(x, ln_g + (l * 3 + 2) * DD, ln_b + (l * 3 + 2) * DD, lnf32);
        cvt(lnf32, lnb, MM * DD);
        gemm(lnb, DD, 0, 0, W1t + (size_t)l * FF * DD, DD, 0, 0,
             ffh, FF, 0, 0, MM, FF, DD, 1, 1, b1 + l * FF, /*bias|relu*/3);
        cvt(ffh, ffhb, MM * FF);
        gemm(ffhb, FF, 0, 0, W2t + (size_t)l * DD * FF, FF, 0, 0,
             hf, DD, 0, 0, MM, DD, FF, 1, 1, b2 + l * DD, /*bias*/2);
        k_add<<<(MM * DD + 255) / 256, 256, 0, stream>>>(x, hf, MM * DD);
    }

    // final LN + weight-tied projection, written directly as preds[T,B,V]
    k_ln<<<MM, 256, 0, stream>>>(x, lnf_g, lnf_b, lnf32);
    cvt(lnf32, lnb, MM * DD);
    gemm(lnb, DD, TT * DD, 0, Eb, DD, 0, 0,
         out, BB * VV, VV, 0,
         TT, VV, DD, 1, BB, nullptr, 0);
}
